// WindowAttention_38895223833008
// MI455X (gfx1250) — matrix operand order repaired and checked by reading, not compiled, not run on hardware
//
#include <hip/hip_runtime.h>
#include <hip/hip_bf16.h>

typedef __attribute__((ext_vector_type(16))) __bf16 v16bf;
typedef __attribute__((ext_vector_type(8)))  float  v8f;

#define WL   49      // window length (7x7)
#define LP   64      // padded length
#define NH   8       // heads
#define HD   32      // head dim
#define EMB  256
#define QK_SCALE 0.17677669529663687f  // 32^-0.5
#define LOG2E 1.4426950408889634f

union ABu { v16bf v; __bf16 e[16]; float4 f4[2]; };
union CDu { v8f   v; float  e[8]; };
union PK4 { __bf16 e[4]; uint2 u; };
union PK8 { __bf16 e[8]; uint4 u; };

// A-operand (16x32 bf16, MxK) from row-major src[ld] at (m0,k0).
__device__ __forceinline__ v16bf ldsA(const __bf16* src, int ld, int m0, int k0, int lane) {
  int m  = m0 + (lane & 15);
  int kh = (lane >> 4) << 3;          // 0 or 8
  const __bf16* p = src + m * ld + k0 + kh;
  ABu a;
  a.f4[0] = *(const float4*)(p);       // K = kh .. kh+7
  a.f4[1] = *(const float4*)(p + 16);  // K = 16+kh .. 16+kh+7
  return a.v;
}

// B-operand (32x16 bf16, KxN) where B[k][n] = src[(n0+n)*ld + k].
__device__ __forceinline__ v16bf ldsBT(const __bf16* src, int ld, int n0, int k0, int lane) {
  int n  = n0 + (lane & 15);
  int kh = (lane >> 4) << 3;          // 0 or 8
  const __bf16* p = src + n * ld + k0 + kh;
  ABu b;
  b.f4[0] = *(const float4*)(p);
  b.f4[1] = *(const float4*)(p + 16);
  return b.v;
}

__device__ __forceinline__ v8f wmma_bf16(v16bf a, v16bf b, v8f c) {
  return __builtin_amdgcn_wmma_f32_16x16x32_bf16(false, a, false, b, (short)0, c, false, false);
}

// Stage a 32 KB tile (bf16, contiguous) global -> LDS via CDNA5 async copy.
__device__ __forceinline__ void stage_async_32k(const __bf16* gsrc, __bf16* ldst, int tid) {
  unsigned lbase = (unsigned)(uintptr_t)ldst + (unsigned)tid * 16u;
  unsigned goff  = (unsigned)tid * 16u;
  unsigned long long gbase = (unsigned long long)(uintptr_t)gsrc;
  #pragma unroll
  for (int i = 0; i < 8; ++i) {
    asm volatile("global_load_async_to_lds_b128 %0, %1, %2"
                 :: "v"(lbase + (unsigned)(i * 4096)),
                    "v"(goff  + (unsigned)(i * 4096)),
                    "s"(gbase)
                 : "memory");
  }
}

__device__ __forceinline__ void wait_async_and_barrier() {
  asm volatile("s_wait_asynccnt 0x0" ::: "memory");
  __syncthreads();
}

// ---------------- pre-kernel: one-time f32 -> bf16 weight conversion into d_ws
__global__ void __launch_bounds__(256)
convert_weights(const float* __restrict__ qkv_w, const float* __restrict__ proj_w,
                __bf16* __restrict__ wq, __bf16* __restrict__ wp) {
  int idx4 = blockIdx.x * 256 + threadIdx.x;      // 65536 float4 total
  const int QKV4 = 768 * 256 / 4;                 // 49152
  float4 v;
  __bf16* dst;
  if (idx4 < QKV4) {
    v = ((const float4*)qkv_w)[idx4];
    dst = wq + idx4 * 4;
  } else {
    v = ((const float4*)proj_w)[idx4 - QKV4];
    dst = wp + (idx4 - QKV4) * 4;
  }
  PK4 p;
  p.e[0] = (__bf16)v.x; p.e[1] = (__bf16)v.y; p.e[2] = (__bf16)v.z; p.e[3] = (__bf16)v.w;
  *(uint2*)dst = p.u;
}

__global__ void __launch_bounds__(256)
swin_window_attn_fused(const float* __restrict__ x,
                       const float* __restrict__ attn_mask,
                       const __bf16* __restrict__ wq_bf,   // [768][256] bf16
                       const float* __restrict__ qkv_b,
                       const __bf16* __restrict__ wp_bf,   // [256][256] bf16
                       const float* __restrict__ proj_b,
                       const float* __restrict__ bias_table,
                       float* __restrict__ out) {
  // 160 KB LDS, region aliasing across stages:
  //  R0: Xbf [64][256]      -> attn_out [64][256]
  //  R1: Wbuf [64][256]     -> P heads 0..3 ([64][64] each)
  //  R2: Qbuf [64][256]     -> P heads 4..7
  //  R3: Kbuf [64][256]
  //  R4: Vt   [256][64]   (v transposed: [dim][token])
  __shared__ __align__(16) unsigned char smem[163840];
  __bf16* Xbf  = (__bf16*)(smem);
  __bf16* Wbuf = (__bf16*)(smem + 32768);
  __bf16* Qbuf = (__bf16*)(smem + 65536);
  __bf16* Kbuf = (__bf16*)(smem + 98304);
  __bf16* Vt   = (__bf16*)(smem + 131072);

  const int b    = blockIdx.x;
  const int tid  = threadIdx.x;
  const int lane = tid & 31;
  const int wid  = tid >> 5;
  const int nloc = lane & 15;
  const int hbase = (lane >> 4) << 3;   // row offset 0 / 8 for C/D layout
  const int mtw  = wid >> 1;            // this wave's fixed M-tile for GEMM stages
  const int ntw0 = (wid & 1) * 2;       // this wave's N-tile pair base

  // ---------------- Stage 0: load x window -> bf16, pad rows 49..63 with 0
  {
    const float4* xs = (const float4*)(x + (size_t)b * WL * EMB);
    #pragma unroll 4
    for (int i = tid; i < LP * EMB / 4; i += 256) {
      int m = i >> 6;                     // row (64 float4 per row)
      float4 v = make_float4(0.f, 0.f, 0.f, 0.f);
      if (m < WL) v = xs[i];
      __bf16* d = Xbf + i * 4;
      d[0] = (__bf16)v.x; d[1] = (__bf16)v.y; d[2] = (__bf16)v.z; d[3] = (__bf16)v.w;
    }
  }
  __syncthreads();

  // ---------------- Stage 1: QKV = Xbf(64x256) @ qkv_w^T(256x768) + b
  // A-tiles (X) invariant across all 12 chunks: load once into registers.
  {
    v16bf a[8];
    #pragma unroll
    for (int ks = 0; ks < 8; ++ks) a[ks] = ldsA(Xbf, EMB, mtw * 16, ks * 32, lane);

    // bias values pipelined one chunk ahead (hide L2 latency under the WMMA chain)
    float bias0 = qkv_b[ntw0 * 16 + nloc];
    float bias1 = qkv_b[ntw0 * 16 + 16 + nloc];

    for (int ch = 0; ch < 12; ++ch) {
      stage_async_32k(wq_bf + (size_t)ch * 64 * EMB, Wbuf, tid);
      float nb0 = bias0, nb1 = bias1;
      if (ch < 11) {
        nb0 = qkv_b[(ch + 1) * 64 + ntw0 * 16 + nloc];
        nb1 = qkv_b[(ch + 1) * 64 + ntw0 * 16 + 16 + nloc];
      }
      wait_async_and_barrier();

      const int sect = ch >> 2;            // 0: Q, 1: K, 2: V  (uniform)
      // two independent accumulator chains (N-tiles ntw0, ntw0+1), interleaved
      CDu acc0, acc1;
      #pragma unroll
      for (int r = 0; r < 8; ++r) { acc0.e[r] = 0.f; acc1.e[r] = 0.f; }
      #pragma unroll
      for (int ks = 0; ks < 8; ++ks) {
        v16bf b0 = ldsBT(Wbuf, EMB, ntw0 * 16,      ks * 32, lane);
        v16bf b1 = ldsBT(Wbuf, EMB, ntw0 * 16 + 16, ks * 32, lane);
        acc0.v = wmma_bf16(a[ks], b0, acc0.v);
        acc1.v = wmma_bf16(a[ks], b1, acc1.v);
      }

      int c0 = (ch & 3) * 64 + ntw0 * 16 + nloc;   // columns within section
      int c1 = c0 + 16;
      int mb = mtw * 16 + hbase;
      if (sect == 0) {
        #pragma unroll
        for (int r = 0; r < 8; ++r) {
          Qbuf[(mb + r) * EMB + c0] = (__bf16)(acc0.e[r] + bias0);
          Qbuf[(mb + r) * EMB + c1] = (__bf16)(acc1.e[r] + bias1);
        }
      } else if (sect == 1) {
        #pragma unroll
        for (int r = 0; r < 8; ++r) {
          Kbuf[(mb + r) * EMB + c0] = (__bf16)(acc0.e[r] + bias0);
          Kbuf[(mb + r) * EMB + c1] = (__bf16)(acc1.e[r] + bias1);
        }
      } else {
        PK8 p0, p1;                                 // 8 consecutive tokens: b128 stores
        #pragma unroll
        for (int r = 0; r < 8; ++r) {
          p0.e[r] = (__bf16)(acc0.e[r] + bias0);
          p1.e[r] = (__bf16)(acc1.e[r] + bias1);
        }
        *(uint4*)(Vt + c0 * LP + mb) = p0.u;
        *(uint4*)(Vt + c1 * LP + mb) = p1.u;
      }
      bias0 = nb0; bias1 = nb1;
      __syncthreads();
    }
  }

  // ---------------- Stage 2: attention, one head per wave
  const int h = wid;
  const __bf16* VtH   = Vt + h * HD * LP;
  const float*  maskW = attn_mask + (size_t)(b & 63) * WL * WL;

  CDu S[4][4];
  #pragma unroll
  for (int mt = 0; mt < 4; ++mt) {
    v16bf aq = ldsA(Qbuf, EMB, mt * 16, h * HD, lane);
    #pragma unroll
    for (int nt = 0; nt < 4; ++nt) {
      v16bf bk = ldsBT(Kbuf, EMB, nt * 16, h * HD, lane);
      CDu c;
      #pragma unroll
      for (int r = 0; r < 8; ++r) c.e[r] = 0.f;
      c.v = wmma_bf16(aq, bk, c.v);
      S[mt][nt] = c;
    }
  }

  // scale + relative-position bias + shifted-window mask (per element of D layout)
  #pragma unroll
  for (int mt = 0; mt < 4; ++mt) {
    #pragma unroll
    for (int nt = 0; nt < 4; ++nt) {
      int n = nt * 16 + nloc;
      #pragma unroll
      for (int r = 0; r < 8; ++r) {
        int m = mt * 16 + hbase + r;
        float v;
        if (m < WL && n < WL) {
          int im = m / 7, jm = m - im * 7;
          int in_ = n / 7, jn = n - in_ * 7;
          int rel = (im - in_ + 6) * 13 + (jm - jn + 6);
          v = S[mt][nt].e[r] * QK_SCALE + bias_table[rel * NH + h] + maskW[m * WL + n];
        } else {
          v = -1e30f;
        }
        S[mt][nt].e[r] = v;
      }
    }
  }

  // row softmax: rows live across 16 lanes of a half-wave (masks 1/2/4/8 stay in-half)
  #pragma unroll
  for (int mt = 0; mt < 4; ++mt) {
    #pragma unroll
    for (int r = 0; r < 8; ++r) {
      float mx = fmaxf(fmaxf(S[mt][0].e[r], S[mt][1].e[r]),
                       fmaxf(S[mt][2].e[r], S[mt][3].e[r]));
      mx = fmaxf(mx, __shfl_xor(mx, 1, 32));
      mx = fmaxf(mx, __shfl_xor(mx, 2, 32));
      mx = fmaxf(mx, __shfl_xor(mx, 4, 32));
      mx = fmaxf(mx, __shfl_xor(mx, 8, 32));
      float sum = 0.f;
      #pragma unroll
      for (int nt = 0; nt < 4; ++nt) {
        float p = exp2f((S[mt][nt].e[r] - mx) * LOG2E);
        S[mt][nt].e[r] = p;
        sum += p;
      }
      sum += __shfl_xor(sum, 1, 32);
      sum += __shfl_xor(sum, 2, 32);
      sum += __shfl_xor(sum, 4, 32);
      sum += __shfl_xor(sum, 8, 32);
      float rinv = 1.f / sum;
      #pragma unroll
      for (int nt = 0; nt < 4; ++nt) S[mt][nt].e[r] *= rinv;
    }
  }

  __syncthreads();  // all waves done reading Qbuf/Wbuf before P overwrites them

  __bf16* Pb = (h < 4) ? (Wbuf + h * LP * LP) : (Qbuf + (h - 4) * LP * LP);
  #pragma unroll
  for (int mt = 0; mt < 4; ++mt)
    #pragma unroll
    for (int nt = 0; nt < 4; ++nt)
      #pragma unroll
      for (int r = 0; r < 8; ++r)
        Pb[(mt * 16 + hbase + r) * LP + nt * 16 + nloc] = (__bf16)S[mt][nt].e[r];

  // O = P(64x64) @ V(64x32) ; attn_out over dead X region
  __bf16* AO = Xbf;
  #pragma unroll
  for (int mt = 0; mt < 4; ++mt) {
    CDu o0, o1;                            // two independent chains (ntv = 0,1)
    #pragma unroll
    for (int r = 0; r < 8; ++r) { o0.e[r] = 0.f; o1.e[r] = 0.f; }
    #pragma unroll
    for (int ks = 0; ks < 2; ++ks) {
      v16bf ap = ldsA (Pb,  LP, mt * 16, ks * 32, lane);
      v16bf v0 = ldsBT(VtH, LP, 0,       ks * 32, lane);
      v16bf v1 = ldsBT(VtH, LP, 16,      ks * 32, lane);
      o0.v = wmma_bf16(ap, v0, o0.v);
      o1.v = wmma_bf16(ap, v1, o1.v);
    }
    int n0 = h * HD + nloc;
    #pragma unroll
    for (int r = 0; r < 8; ++r) {
      AO[(mt * 16 + hbase + r) * EMB + n0]      = (__bf16)o0.e[r];
      AO[(mt * 16 + hbase + r) * EMB + n0 + 16] = (__bf16)o1.e[r];
    }
  }
  __syncthreads();

  // ---------------- Stage 3: out = attn_out(64x256) @ proj_w^T(256x256) + b
  {
    v16bf a[8];
    #pragma unroll
    for (int ks = 0; ks < 8; ++ks) a[ks] = ldsA(AO, EMB, mtw * 16, ks * 32, lane);

    float bias0 = proj_b[ntw0 * 16 + nloc];
    float bias1 = proj_b[ntw0 * 16 + 16 + nloc];

    for (int ch = 0; ch < 4; ++ch) {
      stage_async_32k(wp_bf + (size_t)ch * 64 * EMB, Wbuf, tid);
      float nb0 = bias0, nb1 = bias1;
      if (ch < 3) {
        nb0 = proj_b[(ch + 1) * 64 + ntw0 * 16 + nloc];
        nb1 = proj_b[(ch + 1) * 64 + ntw0 * 16 + 16 + nloc];
      }
      wait_async_and_barrier();

      CDu acc0, acc1;
      #pragma unroll
      for (int r = 0; r < 8; ++r) { acc0.e[r] = 0.f; acc1.e[r] = 0.f; }
      #pragma unroll
      for (int ks = 0; ks < 8; ++ks) {
        v16bf b0 = ldsBT(Wbuf, EMB, ntw0 * 16,      ks * 32, lane);
        v16bf b1 = ldsBT(Wbuf, EMB, ntw0 * 16 + 16, ks * 32, lane);
        acc0.v = wmma_bf16(a[ks], b0, acc0.v);
        acc1.v = wmma_bf16(a[ks], b1, acc1.v);
      }

      int ng0 = ch * 64 + ntw0 * 16 + nloc;
      float* op = out + (size_t)b * WL * EMB;
      #pragma unroll
      for (int r = 0; r < 8; ++r) {
        int m = mtw * 16 + hbase + r;
        if (m < WL) {
          op[m * EMB + ng0]      = acc0.e[r] + bias0;
          op[m * EMB + ng0 + 16] = acc1.e[r] + bias1;
        }
      }
      bias0 = nb0; bias1 = nb1;
      __syncthreads();
    }
  }
}

extern "C" void kernel_launch(void* const* d_in, const int* in_sizes, int n_in,
                              void* d_out, int out_size, void* d_ws, size_t ws_size,
                              hipStream_t stream) {
  (void)in_sizes; (void)n_in; (void)ws_size; (void)out_size;
  const float* x          = (const float*)d_in[0];
  const float* attn_mask  = (const float*)d_in[1];
  const float* qkv_w      = (const float*)d_in[2];
  const float* qkv_b      = (const float*)d_in[3];
  const float* proj_w     = (const float*)d_in[4];
  const float* proj_b     = (const float*)d_in[5];
  const float* bias_table = (const float*)d_in[6];

  __bf16* wq = (__bf16*)d_ws;                // 768*256 bf16 = 384 KB
  __bf16* wp = wq + 768 * 256;               // 256*256 bf16 = 128 KB

  convert_weights<<<256, 256, 0, stream>>>(qkv_w, proj_w, wq, wp);
  swin_window_attn_fused<<<4096, 256, 0, stream>>>(
      x, attn_mask, wq, qkv_b, wp, proj_b, bias_table, (float*)d_out);
}
